// CharLSTM_7327214207307
// MI455X (gfx1250) — matrix operand order repaired and checked by reading, not compiled, not run on hardware
//
#include <hip/hip_runtime.h>

// ---------------------------------------------------------------------------
// CharLSTM on MI455X (gfx1250), wave32 + WMMA bf16 (f32 accumulate).
//
// Shapes: B=64, T=256, H=1024, V=128.  ~554 GFLOP total; weights live in L2
// (192MB), so the sequential scan (T=256 dependent steps/layer) dominates.
// Per-step fused WMMA kernel + batched input-projection GEMMs.
//
// This revision stages the block-shared B (weight) K-chunks in LDS
// (double-buffered, conflict-free 80B row pitch) so the 4 waves of a block
// stop fetching the same tiles 4x from VMEM.
// ---------------------------------------------------------------------------

typedef __bf16 v16bf __attribute__((ext_vector_type(16)));
typedef __bf16 v8bf  __attribute__((ext_vector_type(8)));
typedef float  v8f   __attribute__((ext_vector_type(8)));

#define BPITCH 40  // LDS row pitch in bf16 elems (80B): banks 20r..20r+3, conflict-free

__device__ __forceinline__ __bf16 to_bf16(float f) {
  unsigned u = __float_as_uint(f);
  u += 0x7FFFu + ((u >> 16) & 1u);           // round-to-nearest-even
  unsigned short s = (unsigned short)(u >> 16);
  __bf16 r;
  __builtin_memcpy(&r, &s, sizeof(r));
  return r;
}

__device__ __forceinline__ float sigmoidf_(float x) {
  return 1.0f / (1.0f + __expf(-x));
}

// A-tile (16x32 bf16, row-major source). Lane L holds row M=L%16,
// K = 2*kh..2*kh+15 (elems 0..15), kh = (lane>=16) ? 8 : 0 (matches B-tile).
// Two contiguous 16B loads per lane.
__device__ __forceinline__ v16bf load_a_tile(const __bf16* rowk, int kh) {
  v8bf lo = *(const v8bf*)(rowk + 2 * kh);
  v8bf hi = *(const v8bf*)(rowk + 2 * kh + 8);
  v16bf a;
#pragma unroll
  for (int e = 0; e < 8; ++e) { a[e] = lo[e]; a[8 + e] = hi[e]; }
  return a;
}

// ---------------------------------------------------------------------------
// C[M,N] = A[M,K](bf16) * W[N,K]^T (bf16) + bias[N]   (f32 out)
// Block = 128 threads = 4 waves; block tile = 64 rows x 64 cols,
// wave tile = 16 rows x 64 cols.  The 64 shared W rows of each K-chunk are
// staged in LDS (double buffered) by all 128 threads (one v16bf each), then
// each wave reads its B-tiles from LDS:
//   B-tile layout (32x16): lane holds col N=lane&15; lanes 0-15 K=0..15,
//   lanes 16-31 K=16..31 -> one contiguous 32B LDS read per lane.
// ---------------------------------------------------------------------------
__global__ __launch_bounds__(128) void gemm_bf16_wmma(
    float* __restrict__ C, const __bf16* __restrict__ A,
    const __bf16* __restrict__ W, const float* __restrict__ bias,
    int M, int N, int K, int ldc) {
  __shared__ __bf16 lsB[2][64 * BPITCH];

  const int lane = threadIdx.x & 31;
  const int wave = threadIdx.x >> 5;
  const int m0 = blockIdx.y * 64 + wave * 16;
  const int n0 = blockIdx.x * 64;
  const int r  = lane & 15;
  const int kh = (lane >> 4) * 8;    // A-tile K sub-offset
  const int kb = (lane >> 4) * 16;   // B-tile K sub-offset

  // Cooperative stage of one 64-row x 32-K chunk of W into LDS buffer `buf`.
  const int srow  = threadIdx.x >> 1;        // 0..63
  const int shalf = (threadIdx.x & 1) * 16;  // 0 or 16 (K sub-chunk)
  auto stage = [&](int k, int buf) {
    const __bf16* src = W + (size_t)(n0 + srow) * K + k + shalf;
    *(v16bf*)&lsB[buf][srow * BPITCH + shalf] = *(const v16bf*)src;
  };

  int cur = 0;
  stage(0, 0);
  __syncthreads();

  v8f acc[4] = {};
  const __bf16* arow = A + (size_t)(m0 + r) * K;
  for (int k = 0; k < K; k += 32) {
    if (k + 32 < K) stage(k + 32, cur ^ 1);   // prefetch next chunk
    v16bf a = load_a_tile(arow + k, kh);
#pragma unroll
    for (int nt = 0; nt < 4; ++nt) {
      v16bf b = *(const v16bf*)&lsB[cur][(nt * 16 + r) * BPITCH + kb];
      acc[nt] = __builtin_amdgcn_wmma_f32_16x16x32_bf16(
          false, a, false, b, (short)0, acc[nt], false, false);
    }
    __syncthreads();   // staging stores visible; reads done before overwrite
    cur ^= 1;
  }

  // C/D layout: elem e of lane -> row = m0 + e + (lane>=16 ? 8 : 0)
  const int mh = (lane >> 4) * 8;
#pragma unroll
  for (int nt = 0; nt < 4; ++nt) {
    int col = n0 + nt * 16 + r;
    float bv = bias ? bias[col] : 0.0f;
#pragma unroll
    for (int e = 0; e < 8; ++e) {
      int row = m0 + mh + e;
      C[(size_t)row * ldc + col] = acc[nt][e] + bv;
    }
  }
}

// ---------------------------------------------------------------------------
// One LSTM timestep: g = gx[:,t,:] + h_in @ Whh^T, then fused gate math.
// Grid: 64 blocks (j-tiles of H), 128 threads = 4 waves (batch tiles, B=64).
// The 4 gate B-tiles per K-chunk are shared by all 4 waves -> staged in LDS
// (rows 0..63 = gate*16 + n).  WMMA C layout puts the same (b,j) in the same
// lane/element of all 4 gate accumulators, so the nonlinearity is per-lane.
// h is double-buffered across steps; c is exclusively owned per lane.
// ---------------------------------------------------------------------------
__global__ __launch_bounds__(128) void lstm_step_wmma(
    const __bf16* __restrict__ h_in, __bf16* __restrict__ h_out,
    const __bf16* __restrict__ Whh, const float* __restrict__ gx,
    float* __restrict__ c, __bf16* __restrict__ y, int t) {
  __shared__ __bf16 lsB[2][64 * BPITCH];

  const int lane = threadIdx.x & 31;
  const int wave = threadIdx.x >> 5;
  const int m0 = wave * 16;         // batch tile base (B=64 -> 4 waves)
  const int j0 = blockIdx.x * 16;   // hidden-column tile base (H=1024 -> 64 blocks)
  const int r  = lane & 15;
  const int kh = (lane >> 4) * 8;
  const int kb = (lane >> 4) * 16;

  // Stage the 4 gate tiles (64 W rows: g*1024 + j0 + n) for one K-chunk.
  const int srow  = threadIdx.x >> 1;        // 0..63 -> gate = srow/16, n = srow%16
  const int shalf = (threadIdx.x & 1) * 16;
  const int sgate = srow >> 4;
  const int sn    = srow & 15;
  auto stage = [&](int k, int buf) {
    const __bf16* src = Whh + (size_t)(sgate * 1024 + j0 + sn) * 1024 + k + shalf;
    *(v16bf*)&lsB[buf][srow * BPITCH + shalf] = *(const v16bf*)src;
  };

  int cur = 0;
  stage(0, 0);
  __syncthreads();

  v8f acc[4] = {};
  const __bf16* arow = h_in + (size_t)(m0 + r) * 1024;
  for (int k = 0; k < 1024; k += 32) {
    if (k + 32 < 1024) stage(k + 32, cur ^ 1);
    v16bf a = load_a_tile(arow + k, kh);
#pragma unroll
    for (int g = 0; g < 4; ++g) {   // gate order: i, f, g, o (PyTorch)
      v16bf b = *(const v16bf*)&lsB[cur][(g * 16 + r) * BPITCH + kb];
      acc[g] = __builtin_amdgcn_wmma_f32_16x16x32_bf16(
          false, a, false, b, (short)0, acc[g], false, false);
    }
    __syncthreads();
    cur ^= 1;
  }

  const int mh = (lane >> 4) * 8;
  const int j = j0 + r;
#pragma unroll
  for (int e = 0; e < 8; ++e) {
    int b = m0 + mh + e;
    size_t grow = ((size_t)b * 256 + t) * 4096;   // gx row = b*T + t
    float gi = acc[0][e] + gx[grow + j];
    float gf = acc[1][e] + gx[grow + 1024 + j];
    float gg = acc[2][e] + gx[grow + 2048 + j];
    float go = acc[3][e] + gx[grow + 3072 + j];
    size_t ci = (size_t)b * 1024 + j;
    float cn = sigmoidf_(gf) * c[ci] + sigmoidf_(gi) * tanhf(gg);
    float hv = sigmoidf_(go) * tanhf(cn);
    c[ci] = cn;
    __bf16 hb = to_bf16(hv);
    h_out[ci] = hb;                                   // carry for next step
    y[((size_t)b * 256 + t) * 1024 + j] = hb;         // layer output (bf16)
  }
}

// ---------------------------------------------------------------------------
// Helpers: f32->bf16 convert, embedding gather, state init.
// ---------------------------------------------------------------------------
__global__ void f32_to_bf16_kernel(const float* __restrict__ in,
                                   __bf16* __restrict__ out, int n) {
  int i = blockIdx.x * blockDim.x + threadIdx.x;
  if (i < n) out[i] = to_bf16(in[i]);
}

__global__ void embed_kernel(const int* __restrict__ x,
                             const float* __restrict__ emb,
                             __bf16* __restrict__ out) {
  int row = blockIdx.x;                 // b*T + t
  int idx = x[row];
  const float* src = emb + (size_t)idx * 1024;
  __bf16* dst = out + (size_t)row * 1024;
  for (int j = threadIdx.x; j < 1024; j += blockDim.x) dst[j] = to_bf16(src[j]);
}

__global__ void init_state_kernel(__bf16* __restrict__ hbuf,
                                  float* __restrict__ c,
                                  const float* __restrict__ h0,
                                  const float* __restrict__ c0) {
  int i = blockIdx.x * blockDim.x + threadIdx.x;
  if (i < 64 * 1024) {
    hbuf[i] = to_bf16(h0[i]);
    c[i] = c0[i];
  }
}

// ---------------------------------------------------------------------------
extern "C" void kernel_launch(void* const* d_in, const int* in_sizes, int n_in,
                              void* d_out, int out_size, void* d_ws, size_t ws_size,
                              hipStream_t stream) {
  (void)in_sizes; (void)n_in; (void)out_size; (void)ws_size;
  const int*   x    = (const int*)d_in[0];
  const float* emb  = (const float*)d_in[1];
  const float* Wih0 = (const float*)d_in[2];
  const float* Whh0 = (const float*)d_in[3];
  const float* b0   = (const float*)d_in[4];
  const float* Wih1 = (const float*)d_in[5];
  const float* Whh1 = (const float*)d_in[6];
  const float* b1   = (const float*)d_in[7];
  const float* fcW  = (const float*)d_in[8];
  const float* fcb  = (const float*)d_in[9];
  const float* h0   = (const float*)d_in[10];
  const float* c0   = (const float*)d_in[11];
  float* out = (float*)d_out;

  const int B = 64, T = 256, H = 1024, V = 128;
  const int BT = B * T;            // 16384
  const int G = 4 * H;             // 4096

  char* ws = (char*)d_ws;
  size_t off = 0;
  auto take = [&](size_t bytes) -> void* {
    void* p = ws + off;
    off = (off + bytes + 255) & ~(size_t)255;
    return p;
  };
  float*  gx     = (float*) take((size_t)BT * G * 4);     // 256 MB (reused per layer)
  __bf16* x0bf   = (__bf16*)take((size_t)BT * H * 2);     // embedded input, bf16
  __bf16* y0bf   = (__bf16*)take((size_t)BT * H * 2);     // layer-0 outputs, bf16
  __bf16* y1bf   = (__bf16*)take((size_t)BT * H * 2);     // layer-1 outputs, bf16
  __bf16* hA     = (__bf16*)take((size_t)B * H * 2);      // h double buffer A
  __bf16* hB     = (__bf16*)take((size_t)B * H * 2);      // h double buffer B
  float*  cbuf   = (float*) take((size_t)B * H * 4);      // cell state (f32)
  __bf16* wih0bf = (__bf16*)take((size_t)G * H * 2);
  __bf16* whh0bf = (__bf16*)take((size_t)G * H * 2);
  __bf16* wih1bf = (__bf16*)take((size_t)G * H * 2);
  __bf16* whh1bf = (__bf16*)take((size_t)G * H * 2);
  __bf16* fcwbf  = (__bf16*)take((size_t)V * H * 2);

  // Convert weights to bf16 (once per launch; deterministic).
  const int nW = G * H;
  f32_to_bf16_kernel<<<(nW + 255) / 256, 256, 0, stream>>>(Wih0, wih0bf, nW);
  f32_to_bf16_kernel<<<(nW + 255) / 256, 256, 0, stream>>>(Whh0, whh0bf, nW);
  f32_to_bf16_kernel<<<(nW + 255) / 256, 256, 0, stream>>>(Wih1, wih1bf, nW);
  f32_to_bf16_kernel<<<(nW + 255) / 256, 256, 0, stream>>>(Whh1, whh1bf, nW);
  f32_to_bf16_kernel<<<(V * H + 255) / 256, 256, 0, stream>>>(fcW, fcwbf, V * H);

  // Embedding gather -> bf16 [BT, H].
  embed_kernel<<<BT, 256, 0, stream>>>(x, emb, x0bf);

  for (int layer = 0; layer < 2; ++layer) {
    const __bf16* Wih = layer ? wih1bf : wih0bf;
    const __bf16* Whh = layer ? whh1bf : whh0bf;
    const float*  bia = layer ? b1 : b0;
    const __bf16* xin = layer ? y0bf : x0bf;
    __bf16*       yout = layer ? y1bf : y0bf;

    // Batched input projection: gx[BT, 4H] = xin @ Wih^T + b.
    gemm_bf16_wmma<<<dim3(G / 64, BT / 64), 128, 0, stream>>>(
        gx, xin, Wih, bia, BT, G, H, G);

    // Initialize h (bf16) and c (f32) from h0/c0 slices.
    init_state_kernel<<<(B * H + 255) / 256, 256, 0, stream>>>(
        hA, cbuf, h0 + (size_t)layer * B * H, c0 + (size_t)layer * B * H);

    // Sequential recurrence, h double-buffered.
    for (int t = 0; t < T; ++t) {
      __bf16* hin  = (t & 1) ? hB : hA;
      __bf16* hout = (t & 1) ? hA : hB;
      lstm_step_wmma<<<H / 16, 128, 0, stream>>>(hin, hout, Whh, gx, cbuf, yout, t);
    }
  }

  // Final FC: out[BT, V] = y1 @ fcW^T + fc_b  (f32 straight to d_out).
  gemm_bf16_wmma<<<dim3(V / 64, BT / 64), 128, 0, stream>>>(
      out, y1bf, fcwbf, fcb, BT, V, H, V);
}
